// MultiHeadAttentionLayer_87368224735274
// MI455X (gfx1250) — matrix operand order repaired and checked by reading, not compiled, not run on hardware
//
#include <hip/hip_runtime.h>
#include <hip/hip_bf16.h>

typedef __attribute__((ext_vector_type(16))) __bf16 v16bf;
typedef __attribute__((ext_vector_type(8)))  float  v8f;
typedef __attribute__((ext_vector_type(4)))  unsigned v4u;
typedef __attribute__((ext_vector_type(8)))  int   v8i;
typedef __attribute__((ext_vector_type(4)))  int   v4i;

constexpr int B_ = 2, S_ = 2048, D_ = 1024, H_ = 16, DK_ = 64;
constexpr int BS_ = B_ * S_;
constexpr float SCALE_ = 0.125f;   // 1/sqrt(DK)
constexpr float LN_EPS_ = 1e-5f;

enum { MODE_BF16 = 0, MODE_VT = 1, MODE_F32_RESID = 2 };

union FragU { v16bf v; uint4 u[2]; };

// A-fragment (16x32, 16-bit): lane<16 -> row=lane, K = 0..7 and 16..23;
// lane>=16 -> row=lane-16, K = 8..15 and 24..31. p -> K=0 of this lane's row.
__device__ inline v16bf load_frag_a(const __bf16* p, int ks) {
  FragU f;
  f.u[0] = *reinterpret_cast<const uint4*>(p + ks * 8);
  f.u[1] = *reinterpret_cast<const uint4*>(p + 16 + ks * 8);
  return f.v;
}
// B-fragment (32x16, 16-bit): lane<16 -> col=lane, K = 0..7 and 16..23;
// lane>=16 -> K = 8..15 and 24..31. p -> K=0 of this lane's column (column contiguous over K).
__device__ inline v16bf load_frag_b(const __bf16* p, int ks) {
  FragU f;
  f.u[0] = *reinterpret_cast<const uint4*>(p + ks * 8);
  f.u[1] = *reinterpret_cast<const uint4*>(p + 16 + ks * 8);
  return f.v;
}

__device__ inline v8f wmma_bf16(v16bf a, v16bf b, v8f c) {
  return __builtin_amdgcn_wmma_f32_16x16x32_bf16(false, a, false, b, (short)0, c,
                                                 false, false);
}

// ---- raw 32-bit LDS byte offset of a generic pointer into __shared__
__device__ inline unsigned lds_offset(const void* p) {
  return (unsigned)(__SIZE_TYPE__)(const __attribute__((address_space(3))) char*)p;
}

// ---- Tensor Data Mover: 2D tile (rows x width bf16 elems, row stride in elems)
// from global to contiguous LDS. D# per CDNA5 ISA ch.8 (group0/group1, 2D mode).
__device__ inline void tdm_load_2d(__bf16* ldsp, const __bf16* gp,
                                   unsigned width, unsigned rows, unsigned stride) {
#if __has_builtin(__builtin_amdgcn_tensor_load_to_lds)
  const unsigned long long ga = (unsigned long long)gp;
  v4u g0;
  g0[0] = 1u;                                   // count=1, user descriptor
  g0[1] = lds_offset(ldsp);                     // lds_addr (bytes)
  g0[2] = (unsigned)(ga & 0xFFFFFFFFu);         // global_addr[31:0]
  g0[3] = (unsigned)((ga >> 32) & 0x01FFFFFFu)  // global_addr[56:32]
          | (2u << 30);                          // type=2 ("image")
  v8i g1;
  g1[0] = (int)(1u << 16);                      // data_size=1 -> 2 bytes
  g1[1] = (int)((width & 0xFFFFu) << 16);       // tensor_dim0[15:0] (barrier addr 0)
  g1[2] = (int)(((width >> 16) & 0xFFFFu) | ((rows & 0xFFFFu) << 16)); // dim0 hi|dim1 lo
  g1[3] = (int)(((rows >> 16) & 0xFFFFu) | ((width & 0xFFFFu) << 16)); // dim1 hi|tile_dim0
  g1[4] = (int)(rows & 0xFFFFu);                // tile_dim1 (tile_dim2=0)
  g1[5] = (int)stride;                          // tensor_dim0_stride[31:0]
  g1[6] = 0;                                    // stride hi, dim1_stride lo
  g1[7] = 0;
  v4i z4 = {0, 0, 0, 0};
  v8i z8 = {0, 0, 0, 0, 0, 0, 0, 0};
  // 6-arg form (clang-23 / therock-10.0 headers): (g0, g1, g2, g3, g4, cpol)
  __builtin_amdgcn_tensor_load_to_lds(g0, g1, z4, z4, z8, 0);
#else
  const int lane = threadIdx.x & 31;
  for (unsigned e = lane; e < width * rows; e += 32)
    ldsp[e] = gp[(size_t)(e / width) * stride + (e % width)];
#endif
}

__device__ inline void wait_tensor0() {
#if __has_builtin(__builtin_amdgcn_s_wait_tensorcnt)
  __builtin_amdgcn_s_wait_tensorcnt(0);
#else
  asm volatile("s_wait_tensorcnt 0x0" ::: "memory");
#endif
  asm volatile("" ::: "memory");
}
__device__ inline void wait_tensor2() {
#if __has_builtin(__builtin_amdgcn_s_wait_tensorcnt)
  __builtin_amdgcn_s_wait_tensorcnt(2);
#else
  asm volatile("s_wait_tensorcnt 0x2" ::: "memory");
#endif
  asm volatile("" ::: "memory");
}

// ---------------------------------------------------------------- converts
__global__ __launch_bounds__(256)
void f32_to_bf16_kernel(const float* __restrict__ src, __bf16* __restrict__ dst, int n) {
  int i = blockIdx.x * blockDim.x + threadIdx.x;
  if (i < n) dst[i] = (__bf16)src[i];
}

__global__ __launch_bounds__(256)
void transpose_f32_to_bf16_kernel(const float* __restrict__ W, __bf16* __restrict__ Wt) {
  int i = blockIdx.x * blockDim.x + threadIdx.x;   // i < D*D
  int rr = i >> 10;
  int cc = i & (D_ - 1);
  Wt[(size_t)cc * D_ + rr] = (__bf16)W[i];
}

// ---------------------------------------------------------------- GEMM
// C[M,N] = A[M,K] @ B[K,N]; A row-major bf16, B transposed (Bt[N,K]).
// Wave computes 16x64. K-loop is unrolled by 2 with ping-pong fragment sets:
// WMMAs on set A overlap the loads of set B and vice versa — no register
// rotation moves, loads always one chunk ahead of the consuming WMMAs.
__global__ __launch_bounds__(128)
void gemm_bf16_wmma(const __bf16* __restrict__ A, const __bf16* __restrict__ Bt,
                    int M, int N, int K, int mode,
                    const float* __restrict__ resid,
                    __bf16* __restrict__ outb, float* __restrict__ outf) {
  const int lane = threadIdx.x & 31;
  const int wid  = (blockIdx.x * blockDim.x + threadIdx.x) >> 5;
  const int ntiles = N >> 6;
  const int tm = wid / ntiles, tn = wid - tm * ntiles;
  const int row0 = tm << 4, col0 = tn << 6;
  const int r = lane & 15, ks = lane >> 4, rb = ks * 8;

  v8f acc[4] = {};
  const __bf16* arow = A + (size_t)(row0 + r) * K;
  const __bf16* brow[4];
#pragma unroll
  for (int t = 0; t < 4; ++t) brow[t] = Bt + (size_t)(col0 + t * 16 + r) * K;

  // set A <- chunk 0
  v16bf aA = load_frag_a(arow, ks);
  v16bf bA[4];
#pragma unroll
  for (int t = 0; t < 4; ++t) bA[t] = load_frag_b(brow[t], ks);

  int k0 = 0;
#pragma unroll 1
  for (; k0 + 64 < K; k0 += 64) {
    // set B <- chunk k0+32 (overlaps WMMAs on set A)
    v16bf aB = load_frag_a(arow + k0 + 32, ks);
    v16bf bB[4];
#pragma unroll
    for (int t = 0; t < 4; ++t) bB[t] = load_frag_b(brow[t] + k0 + 32, ks);
#pragma unroll
    for (int t = 0; t < 4; ++t) acc[t] = wmma_bf16(aA, bA[t], acc[t]);
    // set A <- chunk k0+64 (overlaps WMMAs on set B)
    aA = load_frag_a(arow + k0 + 64, ks);
#pragma unroll
    for (int t = 0; t < 4; ++t) bA[t] = load_frag_b(brow[t] + k0 + 64, ks);
#pragma unroll
    for (int t = 0; t < 4; ++t) acc[t] = wmma_bf16(aB, bB[t], acc[t]);
  }
  // epilogue: set A holds chunk k0; load final chunk k0+32 into set B
  {
    v16bf aB = load_frag_a(arow + k0 + 32, ks);
    v16bf bB[4];
#pragma unroll
    for (int t = 0; t < 4; ++t) bB[t] = load_frag_b(brow[t] + k0 + 32, ks);
#pragma unroll
    for (int t = 0; t < 4; ++t) acc[t] = wmma_bf16(aA, bA[t], acc[t]);
#pragma unroll
    for (int t = 0; t < 4; ++t) acc[t] = wmma_bf16(aB, bB[t], acc[t]);
  }

#pragma unroll
  for (int t = 0; t < 4; ++t) {
#pragma unroll
    for (int i = 0; i < 8; ++i) {
      const int row = row0 + rb + i;
      const int col = col0 + t * 16 + r;
      const float v = acc[t][i];
      if (mode == MODE_BF16) {
        outb[(size_t)row * N + col] = (__bf16)v;
      } else if (mode == MODE_VT) {
        const int bb = row / S_, ss = row - bb * S_;            // N == D here
        outb[((size_t)bb * N + col) * S_ + ss] = (__bf16)v;     // Vt[b][d][s]
      } else {
        const size_t idx = (size_t)row * N + col;
        outf[idx] = v + resid[idx];
      }
    }
  }
}

// ---------------------------------------------------------------- attention
// One wave per (b, h, 16-query) tile, 2 waves/block. K/V 32-key chunks are
// staged into LDS by the Tensor Data Mover, double-buffered and overlapped
// with WMMA compute (s_wait_tensorcnt 2 waits only on the current buffer).
__global__ __launch_bounds__(64)
void attn_wmma(const __bf16* __restrict__ Qb, const __bf16* __restrict__ Kb,
               const __bf16* __restrict__ Vt, const int* __restrict__ mask,
               __bf16* __restrict__ ctx) {
  const int lane = threadIdx.x & 31;
  const int w = threadIdx.x >> 5;
  const int wid = blockIdx.x * 2 + w;
  const int qt = wid & (S_ / 16 - 1);
  const int h  = (wid / (S_ / 16)) & (H_ - 1);
  const int b  = wid / ((S_ / 16) * H_);
  const int q0 = qt << 4;
  const int r = lane & 15, ks = lane >> 4, rb = ks * 8;

  // per-wave tiles: K chunk 32x64, V chunk 64x32 (both dbl-buffered), P 16x32
  __shared__ __attribute__((aligned(16))) __bf16 kbuf[2][2][32 * 64];
  __shared__ __attribute__((aligned(16))) __bf16 vbuf[2][2][64 * 32];
  __shared__ __attribute__((aligned(16))) __bf16 pbuf[2][16 * 32];
  __bf16* P = &pbuf[w][0];

  const __bf16* kgbase = Kb + ((size_t)b * S_ * D_ + h * DK_);   // + kc*D_
  const __bf16* vgbase = Vt + ((size_t)b * D_ + h * DK_) * S_;   // + kc

  const __bf16* qrow = Qb + ((size_t)(b * S_ + q0 + r) * D_ + h * DK_);
  const v16bf qa0 = load_frag_a(qrow, ks);
  const v16bf qa1 = load_frag_a(qrow + 32, ks);

  v8f acc[4] = {};
  float mrow[8], lrow[8];
#pragma unroll
  for (int i = 0; i < 8; ++i) { mrow[i] = -3.0e38f; lrow[i] = 0.f; }

  const int* mbase = mask + (size_t)b * S_ * S_;

  // prologue: stage chunk 0 into buffer 0
  tdm_load_2d(&kbuf[w][0][0], kgbase, 64, 32, D_);
  tdm_load_2d(&vbuf[w][0][0], vgbase, 32, 64, S_);
  int cur = 0;

#pragma unroll 1
  for (int kc = 0; kc < S_; kc += 32) {
    if (kc + 32 < S_) {                       // prefetch next chunk
      tdm_load_2d(&kbuf[w][cur ^ 1][0], kgbase + (size_t)(kc + 32) * D_, 64, 32, D_);
      tdm_load_2d(&vbuf[w][cur ^ 1][0], vgbase + (kc + 32), 32, 64, S_);
      wait_tensor2();                         // current buffer (first 2 ops) done
    } else {
      wait_tensor0();
    }
    const __bf16* kb = &kbuf[w][cur][0];
    const __bf16* vb = &vbuf[w][cur][0];

    // ---- scores: S16x32 = q(16x64) @ k^T, two 16x16 C frags, from LDS
    v8f s0 = {}, s1 = {};
    s0 = wmma_bf16(qa0, load_frag_b(kb + r * 64, ks), s0);
    s0 = wmma_bf16(qa1, load_frag_b(kb + r * 64 + 32, ks), s0);
    s1 = wmma_bf16(qa0, load_frag_b(kb + (16 + r) * 64, ks), s1);
    s1 = wmma_bf16(qa1, load_frag_b(kb + (16 + r) * 64 + 32, ks), s1);

    // ---- scale + mask (mask!=0 -> -1e9)
    float v0[8], v1[8];
    const int key0 = kc + r, key1 = kc + 16 + r;
#pragma unroll
    for (int i = 0; i < 8; ++i) {
      const int qq = q0 + rb + i;
      const int* mr = mbase + (size_t)qq * S_;
      v0[i] = mr[key0] ? -1e9f : s0[i] * SCALE_;
      v1[i] = mr[key1] ? -1e9f : s1[i] * SCALE_;
    }

    // ---- online softmax (row lives across the 16 lanes of each half-wave)
#pragma unroll
    for (int i = 0; i < 8; ++i) {
      float mx = fmaxf(v0[i], v1[i]);
#pragma unroll
      for (int off = 8; off; off >>= 1) mx = fmaxf(mx, __shfl_xor(mx, off, 32));
      const float mnew  = fmaxf(mrow[i], mx);
      const float alpha = __expf(mrow[i] - mnew);
      const float p0 = __expf(v0[i] - mnew);
      const float p1 = __expf(v1[i] - mnew);
      float rs = p0 + p1;
#pragma unroll
      for (int off = 8; off; off >>= 1) rs += __shfl_xor(rs, off, 32);
      lrow[i] = lrow[i] * alpha + rs;
      mrow[i] = mnew;
#pragma unroll
      for (int t = 0; t < 4; ++t) acc[t][i] *= alpha;
      P[(rb + i) * 32 + r]      = (__bf16)p0;   // stage P (C-layout) into LDS
      P[(rb + i) * 32 + 16 + r] = (__bf16)p1;
    }
    asm volatile("s_wait_dscnt 0" ::: "memory");

    // ---- ctx += P(16x32) @ V(32x64), V fragments from LDS
    const v16bf pa = load_frag_a(P + r * 32, ks);
#pragma unroll
    for (int t = 0; t < 4; ++t)
      acc[t] = wmma_bf16(pa, load_frag_b(vb + (t * 16 + r) * 32, ks), acc[t]);

    cur ^= 1;
  }

  // ---- normalize by row sum, store ctx bf16 [B,S,D]
#pragma unroll
  for (int t = 0; t < 4; ++t) {
#pragma unroll
    for (int i = 0; i < 8; ++i) {
      const float inv = lrow[i] > 0.f ? 1.0f / lrow[i] : 0.f;
      const int row = q0 + rb + i;
      const int col = h * DK_ + t * 16 + r;
      ctx[(size_t)(b * S_ + row) * D_ + col] = (__bf16)(acc[t][i] * inv);
    }
  }
}

// ---------------------------------------------------------------- layernorm
__global__ __launch_bounds__(256)
void ln_kernel(const float* __restrict__ x, const float* __restrict__ gamma,
               const float* __restrict__ beta, float* __restrict__ out) {
  const int row = blockIdx.x;
  const int t = threadIdx.x;
  const float* xr = x + (size_t)row * D_;
  float s = 0.f, s2 = 0.f;
  for (int c = t; c < D_; c += 256) { const float v = xr[c]; s += v; s2 += v * v; }
#pragma unroll
  for (int off = 16; off; off >>= 1) {
    s  += __shfl_xor(s, off, 32);
    s2 += __shfl_xor(s2, off, 32);
  }
  __shared__ float sa[8], sb[8];
  if ((t & 31) == 0) { sa[t >> 5] = s; sb[t >> 5] = s2; }
  __syncthreads();
  float S = 0.f, S2 = 0.f;
#pragma unroll
  for (int i = 0; i < 8; ++i) { S += sa[i]; S2 += sb[i]; }
  const float mu  = S * (1.0f / D_);
  const float var = S2 * (1.0f / D_) - mu * mu;
  const float inv = rsqrtf(var + LN_EPS_);
  for (int c = t; c < D_; c += 256)
    out[(size_t)row * D_ + c] = (xr[c] - mu) * inv * gamma[c] + beta[c];
}

// ---------------------------------------------------------------- launch
extern "C" void kernel_launch(void* const* d_in, const int* in_sizes, int n_in,
                              void* d_out, int out_size, void* d_ws, size_t ws_size,
                              hipStream_t stream) {
  (void)in_sizes; (void)n_in; (void)out_size; (void)ws_size;

  const float* inQ  = (const float*)d_in[0];
  const float* inK  = (const float*)d_in[1];
  const float* inV  = (const float*)d_in[2];
  const int*   mask = (const int*)d_in[3];
  const float* WQ   = (const float*)d_in[4];
  const float* WK   = (const float*)d_in[5];
  const float* WV   = (const float*)d_in[6];
  const float* WO   = (const float*)d_in[7];
  const float* gamma = (const float*)d_in[8];
  const float* beta  = (const float*)d_in[9];

  char* ws = (char*)d_ws;
  constexpr size_t SZX = (size_t)BS_ * D_ * sizeof(__bf16);  // 8 MB
  constexpr size_t SZW = (size_t)D_ * D_ * sizeof(__bf16);   // 2 MB
  size_t o = 0;
  __bf16* Xq  = (__bf16*)(ws + o); o += SZX;
  __bf16* Xk  = (__bf16*)(ws + o); o += SZX;
  __bf16* Xv  = (__bf16*)(ws + o); o += SZX;
  __bf16* WtQ = (__bf16*)(ws + o); o += SZW;
  __bf16* WtK = (__bf16*)(ws + o); o += SZW;
  __bf16* WtV = (__bf16*)(ws + o); o += SZW;
  __bf16* WtO = (__bf16*)(ws + o); o += SZW;
  __bf16* Qb  = (__bf16*)(ws + o); o += SZX;
  __bf16* Kb  = (__bf16*)(ws + o); o += SZX;
  __bf16* Vt  = (__bf16*)(ws + o); o += SZX;
  __bf16* Cx  = (__bf16*)(ws + o); o += SZX;
  float*  pre = (float*)(ws + o);  o += (size_t)BS_ * D_ * sizeof(float);

  const int n = BS_ * D_;
  f32_to_bf16_kernel<<<n / 256, 256, 0, stream>>>(inQ, Xq, n);
  f32_to_bf16_kernel<<<n / 256, 256, 0, stream>>>(inK, Xk, n);
  f32_to_bf16_kernel<<<n / 256, 256, 0, stream>>>(inV, Xv, n);
  transpose_f32_to_bf16_kernel<<<(D_ * D_) / 256, 256, 0, stream>>>(WQ, WtQ);
  transpose_f32_to_bf16_kernel<<<(D_ * D_) / 256, 256, 0, stream>>>(WK, WtK);
  transpose_f32_to_bf16_kernel<<<(D_ * D_) / 256, 256, 0, stream>>>(WV, WtV);
  transpose_f32_to_bf16_kernel<<<(D_ * D_) / 256, 256, 0, stream>>>(WO, WtO);

  const int gemmBlocks = (BS_ / 16) * (D_ / 64) / 4;   // 1024 blocks of 4 waves
  gemm_bf16_wmma<<<gemmBlocks, 128, 0, stream>>>(Xq, WtQ, BS_, D_, D_, MODE_BF16,
                                                 nullptr, Qb, nullptr);
  gemm_bf16_wmma<<<gemmBlocks, 128, 0, stream>>>(Xk, WtK, BS_, D_, D_, MODE_BF16,
                                                 nullptr, Kb, nullptr);
  gemm_bf16_wmma<<<gemmBlocks, 128, 0, stream>>>(Xv, WtV, BS_, D_, D_, MODE_VT,
                                                 nullptr, Vt, nullptr);

  const int attnBlocks = (B_ * H_ * (S_ / 16)) / 2;    // 2048 blocks of 2 waves
  attn_wmma<<<attnBlocks, 64, 0, stream>>>(Qb, Kb, Vt, mask, Cx);

  gemm_bf16_wmma<<<gemmBlocks, 128, 0, stream>>>(Cx, WtO, BS_, D_, D_, MODE_F32_RESID,
                                                 inQ, nullptr, pre);

  ln_kernel<<<BS_, 256, 0, stream>>>(pre, gamma, beta, (float*)d_out);
}
